// COTIC_81192061763733
// MI455X (gfx1250) — compile-verified, matrix operand order checked by reading, not run on hardware
//
#include <hip/hip_runtime.h>

typedef __attribute__((ext_vector_type(16))) __bf16 v16bf;
typedef __attribute__((ext_vector_type(8)))  __bf16 v8bf;
typedef __attribute__((ext_vector_type(2)))  __bf16 v2bf;
typedef __attribute__((ext_vector_type(8)))  float  v8f;

#define BATCH 16
#define SEQ   2048
#define NROWS (BATCH * SEQ)      // 32768
#define COUT  128
#define KTAPS 5
#define H1N   8
#define H2N   16
#define H3N   32
#define HEFF  34                  // 32 MLP channels + b4 channel + skip channel
#define HCH   8                   // h-channels packed per LDS chunk
#define NCHUNK (H3N / HCH)        // 4 full chunks, then 2-channel tail
#define ROWS_PER_TILE 16
#define BN_EPS 1e-5f

// ---------------------------------------------------------------------------
// Embedding gather: feat0[row, i] = emb[type[row], i], Cin = 64
// ---------------------------------------------------------------------------
__global__ void cotic_embed_kernel(const int* __restrict__ types,
                                   const float* __restrict__ emb,
                                   float* __restrict__ feat0) {
    int idx = blockIdx.x * 256 + threadIdx.x;
    if (idx >= NROWS * 64) return;
    int i = idx & 63;
    int row = idx >> 6;
    feat0[idx] = emb[types[row] * 64 + i];
}

// ---------------------------------------------------------------------------
// Pack W4 / b4 / Wskip into WMMA B-fragment layout (bf16):
// packW[((h*CC + c)*8 + nb)*32 + lane][16 halves], 32x16 tile per (h,c,nb).
// Lanes 0-15 -> n=lane, K in {0..7}U{16..23}; lanes 16-31 -> n=lane-16,
// K in {8..15}U{24..31}; VGPR v holds K pair (2v, 2v+1) (+16 for v>=4).
// ---------------------------------------------------------------------------
__global__ void cotic_pack_kernel(const float* __restrict__ W4,
                                  const float* __restrict__ b4,
                                  const float* __restrict__ Wskip,
                                  __bf16* __restrict__ packW, int cin) {
    int cc = cin / 32;
    int total = HEFF * cc * 8 * 32 * 16;
    int e = blockIdx.x * 256 + threadIdx.x;
    if (e >= total) return;
    int s    = e & 15;
    int lane = (e >> 4) & 31;
    int nb   = (e >> 9) & 7;
    int c    = (e >> 12) % cc;
    int h    = e / (4096 * cc);
    int v = s >> 1, hf = s & 1;
    int Kloc = ((v < 4) ? (2 * v + hf) : (16 + 2 * (v - 4) + hf)) + ((lane < 16) ? 0 : 8);
    int n = lane & 15;
    int i = 32 * c + Kloc;
    int o = nb * 16 + n;
    float val;
    if (h < H3N)        val = W4[((size_t)h * cin + i) * COUT + o];
    else if (h == H3N)  val = b4[(size_t)i * COUT + o];
    else                val = Wskip[(size_t)i * COUT + o];
    packW[e] = (__bf16)val;
}

// ---------------------------------------------------------------------------
__global__ void cotic_zero_stats(float* __restrict__ stats) {
    stats[threadIdx.x] = 0.0f;   // 256 threads: [0..127]=sum, [128..255]=sumsq
}

__global__ void cotic_finalize_stats(float* __restrict__ stats,
                                     const float* __restrict__ gamma,
                                     const float* __restrict__ beta) {
    int c = threadIdx.x;          // 128 threads
    float inv_n = 1.0f / (float)NROWS;
    float mean = stats[c] * inv_n;
    float var  = stats[COUT + c] * inv_n - mean * mean;
    float scale = gamma[c] * rsqrtf(var + BN_EPS);
    stats[c]        = scale;                    // in-place: scale
    stats[COUT + c] = beta[c] - mean * scale;   // shift
}

__global__ void cotic_apply_kernel(const float* __restrict__ src,
                                   const float* __restrict__ stats,
                                   float* __restrict__ dst) {
    int idx = blockIdx.x * 256 + threadIdx.x;
    if (idx >= NROWS * COUT) return;
    int c = idx & (COUT - 1);
    float y = stats[c] * src[idx] + stats[COUT + c];
    dst[idx] = (y >= 0.0f) ? y : 0.1f * y;
}

// ---------------------------------------------------------------------------
// Main fused continuous-conv layer.
// One workgroup (8 waves, wave32) per 16-row tile of (b,l) positions.
// Wave w owns output columns [16w, 16w+16).
// Chunked h-loop: build HCH=8 channel slices M_h[16][HCH*CIN] bf16 per sync
// phase (branch-free), then each wave runs HCH*CC back-to-back WMMAs.
// Tail phase handles the b4 channel (h=32) and the skip channel (h=33).
// ---------------------------------------------------------------------------
template <int CIN>
__global__ __launch_bounds__(256)
void cotic_conv_kernel(const float* __restrict__ times,
                       const float* __restrict__ feat,   // [NROWS, CIN]
                       const float* __restrict__ W1, const float* __restrict__ b1,
                       const float* __restrict__ W2, const float* __restrict__ b2,
                       const float* __restrict__ W3, const float* __restrict__ b3,
                       const __bf16* __restrict__ packW, // [HEFF][CC][8][32][16] bf16
                       const float* __restrict__ bskip,
                       float* __restrict__ outPre,       // [NROWS, COUT]
                       float* __restrict__ stats,        // [2][COUT]
                       int dilation) {
    constexpr int CC  = CIN / 32;
    constexpr int CH  = CIN / 2;   // pairs per row per channel
    __shared__ float  s_f[ROWS_PER_TILE][KTAPS + 1][CIN]; // taps 0..4 lagged, 5 = current
    __shared__ float  s_u[ROWS_PER_TILE][KTAPS][H3N];
    __shared__ __attribute__((aligned(16))) __bf16 s_m[ROWS_PER_TILE][HCH * CIN];

    const int tid  = threadIdx.x;
    const int lane = tid & 31;
    const int wave = tid >> 5;            // 0..7 -> output col block
    const int r0   = blockIdx.x * ROWS_PER_TILE;
    const int b    = r0 / SEQ;
    const int l0   = r0 % SEQ;

    // ---- stage tap features into LDS (zero when lag runs off the front) ----
    const int FTOT = ROWS_PER_TILE * (KTAPS + 1) * CIN;
    for (int e = tid; e < FTOT; e += 256) {
        int i   = e & (CIN - 1);
        int rt  = e / CIN;
        int tk  = rt % (KTAPS + 1);
        int row = rt / (KTAPS + 1);
        int lag = (tk < KTAPS) ? (KTAPS - tk) * dilation : 0;
        int lsrc = l0 + row - lag;
        s_f[row][tk][i] = (lsrc >= 0) ? feat[((size_t)b * SEQ + lsrc) * CIN + i] : 0.0f;
    }

    // ---- kernel-MLP on delta-t: one thread per (row, tap) ----
    if (tid < ROWS_PER_TILE * KTAPS) {
        int row = tid / KTAPS, k = tid % KTAPS;
        int lag = (KTAPS - k) * dilation;
        int l = l0 + row;
        float dt = 0.0f;
        if (l - lag >= 0)
            dt = times[(size_t)b * SEQ + l] - times[(size_t)b * SEQ + l - lag];
        float a1[H1N];
        #pragma unroll
        for (int j = 0; j < H1N; ++j) { float t = dt * W1[j] + b1[j]; a1[j] = t > 0.f ? t : 0.f; }
        float a2[H2N];
        for (int j = 0; j < H2N; ++j) {
            float t = b2[j];
            #pragma unroll
            for (int p = 0; p < H1N; ++p) t += a1[p] * W2[p * H2N + j];
            a2[j] = t > 0.f ? t : 0.f;
        }
        for (int j = 0; j < H3N; ++j) {
            float t = b3[j];
            #pragma unroll
            for (int p = 0; p < H2N; ++p) t += a2[p] * W3[p * H3N + j];
            s_u[row][k][j] = t > 0.f ? t : 0.f;
        }
    }
    __syncthreads();

    v8f acc = {};
    const int mrow  = lane & 15;
    const int kbase = (lane < 16) ? 0 : 8;

    // ---- phase 1: 4 chunks of 8 MLP channels each (branch-free build) ----
    for (int ch = 0; ch < NCHUNK; ++ch) {
        const int hbase = ch * HCH;
        // build M slab: pair of adjacent i per thread -> packed bf16x2 store
        for (int e = tid; e < ROWS_PER_TILE * HCH * CH; e += 256) {
            int ip  = (e & (CH - 1)) * 2;
            int hl  = (e / CH) & (HCH - 1);
            int row = e / (CH * HCH);
            const float* uf = &s_u[row][0][hbase + hl];
            float m0 = 0.f, m1 = 0.f;
            #pragma unroll
            for (int k = 0; k < KTAPS; ++k) {
                float u = uf[k * H3N];
                m0 += u * s_f[row][k][ip];
                m1 += u * s_f[row][k][ip + 1];
            }
            v2bf pk; pk[0] = (__bf16)m0; pk[1] = (__bf16)m1;
            *(v2bf*)&s_m[row][hl * CIN + ip] = pk;
        }
        __syncthreads();

        // consume: HCH * CC WMMAs per wave, fully unrolled
        #pragma unroll
        for (int hl = 0; hl < HCH; ++hl) {
            #pragma unroll
            for (int c = 0; c < CC; ++c) {
                const __bf16* mp = &s_m[mrow][hl * CIN + c * 32 + kbase];
                v8bf alo = *(const v8bf*)mp;
                v8bf ahi = *(const v8bf*)(mp + 16);
                v16bf A = __builtin_shufflevector(alo, ahi,
                            0, 1, 2, 3, 4, 5, 6, 7, 8, 9, 10, 11, 12, 13, 14, 15);
                const __bf16* pb = packW +
                    ((size_t)(((hbase + hl) * CC + c) * 8 + wave) * 32 + lane) * 16;
                v16bf Bm = *(const v16bf*)pb;
                acc = __builtin_amdgcn_wmma_f32_16x16x32_bf16(
                          false, A, false, Bm, (short)0, acc, false, false);
            }
        }
        __syncthreads();
    }

    // ---- phase 2: tail channels h=32 (b4: sum of taps), h=33 (skip) ----
    for (int e = tid; e < ROWS_PER_TILE * 2 * CH; e += 256) {
        int ip  = (e & (CH - 1)) * 2;
        int hl  = (e / CH) & 1;
        int row = e / (CH * 2);
        float m0, m1;
        if (hl == 0) {
            m0 = 0.f; m1 = 0.f;
            #pragma unroll
            for (int k = 0; k < KTAPS; ++k) {
                m0 += s_f[row][k][ip];
                m1 += s_f[row][k][ip + 1];
            }
        } else {
            m0 = s_f[row][KTAPS][ip];
            m1 = s_f[row][KTAPS][ip + 1];
        }
        v2bf pk; pk[0] = (__bf16)m0; pk[1] = (__bf16)m1;
        *(v2bf*)&s_m[row][hl * CIN + ip] = pk;
    }
    __syncthreads();
    #pragma unroll
    for (int hl = 0; hl < 2; ++hl) {
        #pragma unroll
        for (int c = 0; c < CC; ++c) {
            const __bf16* mp = &s_m[mrow][hl * CIN + c * 32 + kbase];
            v8bf alo = *(const v8bf*)mp;
            v8bf ahi = *(const v8bf*)(mp + 16);
            v16bf A = __builtin_shufflevector(alo, ahi,
                        0, 1, 2, 3, 4, 5, 6, 7, 8, 9, 10, 11, 12, 13, 14, 15);
            const __bf16* pb = packW +
                ((size_t)(((H3N + hl) * CC + c) * 8 + wave) * 32 + lane) * 16;
            v16bf Bm = *(const v16bf*)pb;
            acc = __builtin_amdgcn_wmma_f32_16x16x32_bf16(
                      false, A, false, Bm, (short)0, acc, false, false);
        }
    }

    // ---- epilogue: +bskip, store pre-BN output, accumulate batch stats ----
    const int col    = wave * 16 + (lane & 15);
    const int rowoff = (lane < 16) ? 0 : 8;
    const float bs = bskip[col];
    float ssum = 0.0f, ssq = 0.0f;
    #pragma unroll
    for (int j = 0; j < 8; ++j) {
        float v = acc[j] + bs;
        outPre[(size_t)(r0 + rowoff + j) * COUT + col] = v;
        ssum += v; ssq += v * v;
    }
    atomicAdd(&stats[col], ssum);
    atomicAdd(&stats[COUT + col], ssq);
}

// ---------------------------------------------------------------------------
extern "C" void kernel_launch(void* const* d_in, const int* in_sizes, int n_in,
                              void* d_out, int out_size, void* d_ws, size_t ws_size,
                              hipStream_t stream) {
    const float* times = (const float*)d_in[0];
    const int*   types = (const int*)d_in[1];
    const float* emb   = (const float*)d_in[2];

    // per-layer params: W1,b1,W2,b2,W3,b3,W4,b4,Wskip,bskip,gamma,beta
    const float* P[4][12];
    for (int i = 0; i < 4; ++i)
        for (int j = 0; j < 12; ++j)
            P[i][j] = (const float*)d_in[3 + 12 * i + j];

    char* ws = (char*)d_ws;
    const size_t BUF = (size_t)NROWS * COUT * sizeof(float);     // 16 MB
    float*  bufA  = (float*)(ws);
    float*  bufB  = (float*)(ws + BUF);
    const size_t PACK_STRIDE = (size_t)HEFF * 4 * 8 * 32 * 16 * sizeof(__bf16); // max CC=4
    __bf16* packW = (__bf16*)(ws + 2 * BUF);
    float*  stats = (float*)(ws + 2 * BUF + 4 * PACK_STRIDE);

    // 1) embedding gather -> bufA (Cin = 64)
    cotic_embed_kernel<<<(NROWS * 64 + 255) / 256, 256, 0, stream>>>(types, emb, bufA);

    // 2) pack per-layer weight tensors into WMMA B-fragment layout
    for (int i = 0; i < 4; ++i) {
        int cin = (i == 0) ? 64 : 128;
        int total = HEFF * (cin / 32) * 8 * 32 * 16;
        cotic_pack_kernel<<<(total + 255) / 256, 256, 0, stream>>>(
            P[i][6], P[i][7], P[i][8],
            (__bf16*)((char*)packW + (size_t)i * PACK_STRIDE), cin);
    }

    // 3) layers (ping-pong bufA/bufB; BN+leaky applied in place, last -> d_out)
    float* cur = bufA;
    float* nxt = bufB;
    for (int i = 0; i < 4; ++i) {
        int dilation = 1 << i;
        __bf16* pw = (__bf16*)((char*)packW + (size_t)i * PACK_STRIDE);
        cotic_zero_stats<<<1, 256, 0, stream>>>(stats);
        if (i == 0) {
            cotic_conv_kernel<64><<<NROWS / ROWS_PER_TILE, 256, 0, stream>>>(
                times, cur, P[i][0], P[i][1], P[i][2], P[i][3], P[i][4], P[i][5],
                pw, P[i][9], nxt, stats, dilation);
        } else {
            cotic_conv_kernel<128><<<NROWS / ROWS_PER_TILE, 256, 0, stream>>>(
                times, cur, P[i][0], P[i][1], P[i][2], P[i][3], P[i][4], P[i][5],
                pw, P[i][9], nxt, stats, dilation);
        }
        cotic_finalize_stats<<<1, 128, 0, stream>>>(stats, P[i][10], P[i][11]);
        float* dst = (i == 3) ? (float*)d_out : nxt;
        cotic_apply_kernel<<<(NROWS * COUT + 255) / 256, 256, 0, stream>>>(nxt, stats, dst);
        // swap: next layer reads the (in-place normalized) output
        float* t = cur; cur = nxt; nxt = t;
    }
    (void)in_sizes; (void)n_in; (void)out_size; (void)ws_size;
}
